// MultiHeadAttention_62440234549615
// MI455X (gfx1250) — matrix operand order repaired and checked by reading, not compiled, not run on hardware
//
#include <hip/hip_runtime.h>

typedef unsigned short u16;
typedef unsigned int   u32;
typedef __attribute__((ext_vector_type(16))) __bf16 v16bf;
typedef __attribute__((ext_vector_type(8)))  float  v8f;
typedef __attribute__((ext_vector_type(4)))  float  v4f;
typedef __attribute__((ext_vector_type(4)))  u32    v4u;

#define D_MODEL 1024
#define NH      16
#define DKH     64
#define SEQ     2048
#define BATCH   4
#define TOKENS  (BATCH * SEQ)   // 8192

// 0.125 (1/sqrt(d_k)) * log2(e): moves softmax into exp2 domain, folded into Q.
#define Q_SCALE 0.18033688011112042f

// ---------- scalar helpers ----------

__device__ __forceinline__ u16 f2bf(float f) {
  u32 u = __float_as_uint(f);
  u32 r = u + 0x7fffu + ((u >> 16) & 1u);   // round-to-nearest-even
  return (u16)(r >> 16);
}

// packed f32x2 -> bf16x2 (one v_cvt_pk_bf16_f32 when available)
__device__ __forceinline__ u32 f2bf_pk(float lo, float hi) {
#if __has_builtin(__builtin_amdgcn_cvt_pk_bf16_f32)
  auto p = __builtin_amdgcn_cvt_pk_bf16_f32(lo, hi);
  u32 r;
  __builtin_memcpy(&r, &p, 4);
  return r;
#else
  return (u32)f2bf(lo) | ((u32)f2bf(hi) << 16);
#endif
}

__device__ __forceinline__ float fast_exp2(float x) {
#if __has_builtin(__builtin_amdgcn_exp2f)
  return __builtin_amdgcn_exp2f(x);
#else
  return exp2f(x);
#endif
}

__device__ __forceinline__ float fast_rcp(float x) {
#if __has_builtin(__builtin_amdgcn_rcpf)
  return __builtin_amdgcn_rcpf(x);
#else
  return 1.0f / x;
#endif
}

// ---------- DPP16 butterfly reductions over 16-lane rows ----------
// ctrl: 0xB1 = quad_perm[1,0,3,2] (xor1), 0x4E = quad_perm[2,3,0,1] (xor2),
//       0x141 = row_half_mirror (pairs across quads), 0x140 = row_mirror.

#if __has_builtin(__builtin_amdgcn_update_dpp)
template <int CTRL>
__device__ __forceinline__ float dpp_f(float x) {
  return __int_as_float(__builtin_amdgcn_update_dpp(
      0, __float_as_int(x), CTRL, 0xf, 0xf, true));
}
__device__ __forceinline__ float row16_max(float v) {
  v = fmaxf(v, dpp_f<0xB1>(v));
  v = fmaxf(v, dpp_f<0x4E>(v));
  v = fmaxf(v, dpp_f<0x141>(v));
  v = fmaxf(v, dpp_f<0x140>(v));
  return v;
}
__device__ __forceinline__ float row16_sum(float v) {
  v += dpp_f<0xB1>(v);
  v += dpp_f<0x4E>(v);
  v += dpp_f<0x141>(v);
  v += dpp_f<0x140>(v);
  return v;
}
#else
__device__ __forceinline__ float row16_max(float v) {
  v = fmaxf(v, __shfl_xor(v, 1, 32));
  v = fmaxf(v, __shfl_xor(v, 2, 32));
  v = fmaxf(v, __shfl_xor(v, 4, 32));
  v = fmaxf(v, __shfl_xor(v, 8, 32));
  return v;
}
__device__ __forceinline__ float row16_sum(float v) {
  v += __shfl_xor(v, 1, 32);
  v += __shfl_xor(v, 2, 32);
  v += __shfl_xor(v, 4, 32);
  v += __shfl_xor(v, 8, 32);
  return v;
}
#endif

// ---------- WMMA helpers ----------

__device__ __forceinline__ v16bf frag2(const u16* p0, const u16* p1) {
  v16bf r;
  v4u* pr = reinterpret_cast<v4u*>(&r);
  pr[0] = *reinterpret_cast<const v4u*>(p0);
  pr[1] = *reinterpret_cast<const v4u*>(p1);
  return r;
}

__device__ __forceinline__ v8f wmma_bf16(v16bf a, v16bf b, v8f c) {
  return __builtin_amdgcn_wmma_f32_16x16x32_bf16(
      /*neg_a=*/false, a, /*neg_b=*/false, b,
      /*c_mod=*/(short)0, c, /*reuse_a=*/false, /*reuse_b=*/false);
}

// ---------- GEMM: Y = (X @ W + b) * out_scale ----------
// Block tile 128(M) x 64(N), K-step 64, 8 waves each computing 32x32.

template <bool A_IS_BF16, bool HEADSPLIT_STORE>
__global__ __launch_bounds__(256) void gemm_proj(
    const void* __restrict__ Ain, const float* __restrict__ W,
    const float* __restrict__ bias, void* __restrict__ Out, float out_scale) {
  constexpr int BM = 128, BN = 64, BK = 64, LDA = BK + 8, LDB = BK + 8;
  __shared__ u16 As[BM * LDA];
  __shared__ u16 Wt[BN * LDB];   // W staged transposed: Wt[n][k]

  const int tid  = threadIdx.x;
  const int lane = tid & 31;
  const int wave = tid >> 5;
  const int wy = wave >> 1, wx = wave & 1;      // 4x2 wave grid
  const int m0 = blockIdx.y * BM, n0 = blockIdx.x * BN;

  v8f acc[2][2] = {};

  for (int k0 = 0; k0 < D_MODEL; k0 += BK) {
    // ---- stage A tile ----
    if (A_IS_BF16) {
      const u16* A = (const u16*)Ain;
#pragma unroll
      for (int j = 0; j < 4; ++j) {
        int c = tid + j * 256;                  // 1024 chunks of 8 u16
        int row = c >> 3, off = (c & 7) * 8;
        *reinterpret_cast<v4u*>(&As[row * LDA + off]) =
            *reinterpret_cast<const v4u*>(&A[(size_t)(m0 + row) * D_MODEL + k0 + off]);
      }
    } else {
      const float* A = (const float*)Ain;
#pragma unroll
      for (int j = 0; j < 8; ++j) {
        int c = tid + j * 256;                  // 2048 v4f chunks
        int row = c >> 4, off = (c & 15) * 4;
        v4f f = __builtin_nontemporal_load(      // activations are read-once
            reinterpret_cast<const v4f*>(&A[(size_t)(m0 + row) * D_MODEL + k0 + off]));
        u32* d = reinterpret_cast<u32*>(&As[row * LDA + off]);
        d[0] = f2bf_pk(f.x, f.y);
        d[1] = f2bf_pk(f.z, f.w);
      }
    }
    // ---- stage W tile transposed, fp32 -> bf16 ----
#pragma unroll
    for (int j = 0; j < 4; ++j) {
      int c = tid + j * 256;                    // 1024 v4f chunks (64x64)
      int kr = c >> 4, off = (c & 15) * 4;
      v4f f = *reinterpret_cast<const v4f*>(
          &W[(size_t)(k0 + kr) * D_MODEL + n0 + off]);
      Wt[(off + 0) * LDB + kr] = f2bf(f.x);
      Wt[(off + 1) * LDB + kr] = f2bf(f.y);
      Wt[(off + 2) * LDB + kr] = f2bf(f.z);
      Wt[(off + 3) * LDB + kr] = f2bf(f.w);
    }
    __syncthreads();

    // ---- WMMA compute ----
    const int ar  = lane & 15, kb  = (lane >> 4) * 8;    // A-frag lane map
    const int bc  = lane & 15, kb2 = (lane >> 4) * 8;    // B-frag lane map
#pragma unroll
    for (int kk = 0; kk < BK; kk += 32) {
      const u16* pa0 = &As[(wy * 32 + ar) * LDA + kk + kb];
      const u16* pa1 = &As[(wy * 32 + 16 + ar) * LDA + kk + kb];
      v16bf a0 = frag2(pa0, pa0 + 16);
      v16bf a1 = frag2(pa1, pa1 + 16);
      const u16* pb0 = &Wt[(wx * 32 + bc) * LDB + kk + kb2];
      const u16* pb1 = &Wt[(wx * 32 + 16 + bc) * LDB + kk + kb2];
      v16bf b0 = frag2(pb0, pb0 + 16);
      v16bf b1 = frag2(pb1, pb1 + 16);
      acc[0][0] = wmma_bf16(a0, b0, acc[0][0]);
      acc[0][1] = wmma_bf16(a0, b1, acc[0][1]);
      acc[1][0] = wmma_bf16(a1, b0, acc[1][0]);
      acc[1][1] = wmma_bf16(a1, b1, acc[1][1]);
    }
    __syncthreads();
  }

  // ---- store C fragments (+bias, *scale) ----
#pragma unroll
  for (int i = 0; i < 2; ++i)
#pragma unroll
    for (int j = 0; j < 2; ++j) {
      int n = n0 + wx * 32 + j * 16 + (lane & 15);
      float bv = bias[n];
#pragma unroll
      for (int r = 0; r < 8; ++r) {
        int m = m0 + wy * 32 + i * 16 + (lane >> 4) * 8 + r;
        float val = (acc[i][j][r] + bv) * out_scale;
        if (HEADSPLIT_STORE) {
          int bb = m >> 11, s = m & (SEQ - 1);
          int h = n >> 6, dk = n & 63;
          ((u16*)Out)[(((size_t)(bb * NH + h)) * SEQ + s) * DKH + dk] = f2bf(val);
        } else {
          __builtin_nontemporal_store(val, &((float*)Out)[(size_t)m * D_MODEL + n]);
        }
      }
    }
}

// ---------- flash attention (online softmax, exp2 domain), WMMA bf16 ----------
// grid: (SEQ/128, B*H), block 256 = 8 waves; each wave owns 16 query rows.
// Q was pre-scaled by 0.125*log2(e), so scores are already in log2 space.

__global__ __launch_bounds__(256) void attn_kernel(
    const u16* __restrict__ Qh, const u16* __restrict__ Kh,
    const u16* __restrict__ Vh, u16* __restrict__ AO) {
  constexpr int LDT = DKH + 8;                 // 72 shorts = 144B rows
  __shared__ u16 Ks[64 * LDT];                 // K tile, row-major [t][d]
  __shared__ u16 Vt[64 * LDT];                 // V tile, transposed [d][t]
  __shared__ u16 Ps[8 * 16 * LDT];             // per-wave P (16x64)

  const int tid = threadIdx.x, lane = tid & 31, wave = tid >> 5;
  const int bh = blockIdx.y;
  const int m0 = blockIdx.x * 128 + wave * 16;
  const u16* Qp = Qh + (size_t)bh * SEQ * DKH;
  const u16* Kp = Kh + (size_t)bh * SEQ * DKH;
  const u16* Vp = Vh + (size_t)bh * SEQ * DKH;

  const int kb  = (lane >> 4) * 8;             // A-frag lane map
  const int bc  = lane & 15;                   // B-frag lane map
  const int kb2 = (lane >> 4) * 8;

  // Q A-fragments, held in VGPRs for the whole kernel
  const size_t qoff = (size_t)(m0 + (lane & 15)) * DKH;
  v16bf qa0 = frag2(&Qp[qoff + kb],      &Qp[qoff + kb + 16]);
  v16bf qa1 = frag2(&Qp[qoff + 32 + kb], &Qp[qoff + 48 + kb]);

  float mrow[8], lrow[8];
  v8f o[4] = {};
#pragma unroll
  for (int r = 0; r < 8; ++r) { mrow[r] = -1e30f; lrow[r] = 0.f; }

  for (int t0 = 0; t0 < SEQ; t0 += 64) {
    // ---- stage K rows (64x64 bf16) ----
#pragma unroll
    for (int j = 0; j < 2; ++j) {
      int c = tid + j * 256;                   // 512 chunks of 8 u16
      int row = c >> 3, off = (c & 7) * 8;
      *reinterpret_cast<v4u*>(&Ks[row * LDT + off]) =
          *reinterpret_cast<const v4u*>(&Kp[(size_t)(t0 + row) * DKH + off]);
    }
    // ---- stage V transposed ----
#pragma unroll
    for (int j = 0; j < 2; ++j) {
      int c = tid + j * 256;
      int row = c >> 3, off = (c & 7) * 8;     // row = t, off = d base
      v4u vv = *reinterpret_cast<const v4u*>(&Vp[(size_t)(t0 + row) * DKH + off]);
      const u16* pv = reinterpret_cast<const u16*>(&vv);
#pragma unroll
      for (int q = 0; q < 8; ++q) Vt[(off + q) * LDT + row] = pv[q];
    }
    if (t0 + 64 < SEQ) {                       // global_prefetch_b8 next tile
      __builtin_prefetch(&Kp[(size_t)(t0 + 64) * DKH], 0, 0);
      __builtin_prefetch(&Vp[(size_t)(t0 + 64) * DKH], 0, 0);
    }
    __syncthreads();

    // ---- scores S = Q K^T (already scaled to log2 domain) ----
    v8f sf[4];
#pragma unroll
    for (int g = 0; g < 4; ++g) {
      const u16* pk0 = &Ks[(g * 16 + bc) * LDT + kb2];        // d = 0..31
      const u16* pk1 = &Ks[(g * 16 + bc) * LDT + 32 + kb2];   // d = 32..63
      v8f z = {};
      sf[g] = wmma_bf16(qa0, frag2(pk0, pk0 + 16), z);
      sf[g] = wmma_bf16(qa1, frag2(pk1, pk1 + 16), sf[g]);
    }

    // ---- online softmax update: DPP butterflies, no LDS traffic ----
#pragma unroll
    for (int r = 0; r < 8; ++r) {
      float v = fmaxf(fmaxf(sf[0][r], sf[1][r]), fmaxf(sf[2][r], sf[3][r]));
      v = row16_max(v);
      float mn = fmaxf(mrow[r], v);
      float sum = 0.f;
#pragma unroll
      for (int g = 0; g < 4; ++g) {
        float p = fast_exp2(sf[g][r] - mn);
        sf[g][r] = p;
        sum += p;
      }
      sum = row16_sum(sum);
      float alpha = fast_exp2(mrow[r] - mn);
      lrow[r] = lrow[r] * alpha + sum;
      mrow[r] = mn;
#pragma unroll
      for (int gd = 0; gd < 4; ++gd) o[gd][r] *= alpha;
    }

    // ---- P: C-layout -> A-layout via per-wave LDS round trip ----
    u16* Pw = &Ps[wave * 16 * LDT];
    {
      int prow = (lane >> 4) * 8, pcol = lane & 15;
#pragma unroll
      for (int g = 0; g < 4; ++g)
#pragma unroll
        for (int r = 0; r < 8; ++r)
          Pw[(prow + r) * LDT + g * 16 + pcol] = f2bf(sf[g][r]);
    }
    asm volatile("s_wait_dscnt 0x0" ::: "memory");
    const u16* Pr = &Ps[wave * 16 * LDT + (lane & 15) * LDT];
    v16bf pa0 = frag2(Pr + kb,      Pr + kb + 16);
    v16bf pa1 = frag2(Pr + 32 + kb, Pr + 48 + kb);

    // ---- O += P V ----
#pragma unroll
    for (int gd = 0; gd < 4; ++gd) {
      const u16* pv0 = &Vt[(gd * 16 + bc) * LDT + kb2];        // t = 0..31
      const u16* pv1 = &Vt[(gd * 16 + bc) * LDT + 32 + kb2];   // t = 32..63
      o[gd] = wmma_bf16(pa0, frag2(pv0, pv0 + 16), o[gd]);
      o[gd] = wmma_bf16(pa1, frag2(pv1, pv1 + 16), o[gd]);
    }
    __syncthreads();
  }

  // ---- epilogue: O /= l, store bf16 merged-head layout [token][1024] ----
  const int b = bh >> 4, h = bh & 15;
#pragma unroll
  for (int r = 0; r < 8; ++r) {
    float inv = fast_rcp(lrow[r]);
    int m = m0 + (lane >> 4) * 8 + r;
#pragma unroll
    for (int gd = 0; gd < 4; ++gd) {
      int col = h * DKH + gd * 16 + (lane & 15);
      AO[((size_t)(b * SEQ + m)) * D_MODEL + col] = f2bf(o[gd][r] * inv);
    }
  }
}

// ---------- launcher ----------

extern "C" void kernel_launch(void* const* d_in, const int* in_sizes, int n_in,
                              void* d_out, int out_size, void* d_ws, size_t ws_size,
                              hipStream_t stream) {
  (void)in_sizes; (void)n_in; (void)out_size; (void)ws_size;
  const float* q  = (const float*)d_in[0];
  const float* k  = (const float*)d_in[1];
  const float* v  = (const float*)d_in[2];
  const float* Wq = (const float*)d_in[3];
  const float* bq = (const float*)d_in[4];
  const float* Wk = (const float*)d_in[5];
  const float* bk = (const float*)d_in[6];
  const float* Wv = (const float*)d_in[7];
  const float* bv = (const float*)d_in[8];
  const float* Wo = (const float*)d_in[9];
  const float* bo = (const float*)d_in[10];

  const size_t HE = (size_t)BATCH * NH * SEQ * DKH;  // 8,388,608 elems (16MB bf16)
  u16* Qh = (u16*)d_ws;
  u16* Kh = Qh + HE;
  u16* Vh = Kh + HE;
  u16* AO = Vh + HE;                                 // total 64MB of d_ws

  dim3 gg(D_MODEL / 64, TOKENS / 128);               // (16, 64)
  gemm_proj<false, true><<<gg, 256, 0, stream>>>(q, Wq, bq, Qh, (float)Q_SCALE);
  gemm_proj<false, true><<<gg, 256, 0, stream>>>(k, Wk, bk, Kh, 1.0f);
  gemm_proj<false, true><<<gg, 256, 0, stream>>>(v, Wv, bv, Vh, 1.0f);

  dim3 ga(SEQ / 128, BATCH * NH);                    // (16, 64)
  attn_kernel<<<ga, 256, 0, stream>>>(Qh, Kh, Vh, AO);

  gemm_proj<true, false><<<gg, 256, 0, stream>>>(AO, Wo, bo, d_out, 1.0f);
}
